// GraphNetBlock_50998441673264
// MI455X (gfx1250) — matrix operand order repaired and checked by reading, not compiled, not run on hardware
//
#include <hip/hip_runtime.h>
#include <hip/hip_bf16.h>
#include <stdint.h>

// ---------------------------------------------------------------------------
// GraphNet block (edge/node/global MLPs with relu+LayerNorm) for gfx1250.
// f16 WMMA (v_wmma_f32_16x16x32_f16) with f32 accumulation.
// ---------------------------------------------------------------------------

typedef __attribute__((ext_vector_type(16))) _Float16 v16h;
typedef __attribute__((ext_vector_type(8)))  float    v8f;
typedef __attribute__((ext_vector_type(4)))  _Float16 h4;

#define B_   8
#define N_   256
#define ND   64      // node dim
#define ED   64      // edge dim
#define GD   32      // global dim
#define HD   128     // hidden
#define LN_EPS 1e-5f

#define WAVE_FENCE() __builtin_amdgcn_fence(__ATOMIC_ACQ_REL, "wavefront")

__device__ __forceinline__ h4 cvt4(const float4 v) {
  h4 r;
  r.x = (_Float16)v.x; r.y = (_Float16)v.y;
  r.z = (_Float16)v.z; r.w = (_Float16)v.w;
  return r;
}

// ---- fragment loads (LDS, 16B vector loads) -------------------------------
__device__ __forceinline__ v16h ld_bfrag(const _Float16* p) {
  union { float4 f[2]; v16h h; } t;
  t.f[0] = *(const float4*)(p);
  t.f[1] = *(const float4*)(p + 8);
  return t.h;
}
__device__ __forceinline__ v16h ld_afrag(const _Float16* p) {
  // A 16-bit layout: halves 0..7 -> K k0..k0+7 ; halves 8..15 -> K k0+16..k0+23
  union { float4 f[2]; v16h h; } t;
  t.f[0] = *(const float4*)(p);
  t.f[1] = *(const float4*)(p + 16);
  return t.h;
}

// ---- stage weights from global f32 into swizzled f16 B-fragment layout ----
// fragment (c, kt) lives at dst + (c*KT + kt)*512 halves; within a fragment,
// lane l owns 16 contiguous halves: element e -> K = kt*32 + (e>>3)*16 + (l>>4)*8 + (e&7),
// N = c*16 + (l&15).   (matches V_WMMA 16x16x32 f16 B layout)
__device__ __forceinline__ void stage_weights(const float* __restrict__ W, int K, int NC,
                                              _Float16* __restrict__ dst, int tid, int nt) {
  const int KT = K >> 5, CT = NC >> 4;
  const int total = KT * CT * 512;
  for (int idx = tid; idx < total; idx += nt) {
    const int frag = idx >> 9, within = idx & 511;
    const int lane = within >> 4, e = within & 15;
    const int c = frag / KT, kt = frag - c * KT;
    const int k = kt * 32 + (e >> 3) * 16 + (lane >> 4) * 8 + (e & 7);
    const int n = c * 16 + (lane & 15);
    dst[idx] = (_Float16)W[k * NC + n];
  }
}

__device__ __forceinline__ void copy_f32(const float* __restrict__ s, float* __restrict__ d,
                                         int n, int tid, int nt) {
  for (int i = tid; i < n; i += nt) d[i] = s[i];
}

// ---- one 16-row GEMM tile: hBuf[16][CT*16] = A(16xK,f16) @ W(KxCT*16) -----
__device__ __forceinline__ void gemm_tile(const _Float16* __restrict__ aBuf, int SA, int KT,
                                          const _Float16* __restrict__ wFrag, int CT,
                                          float* __restrict__ hBuf, int lane) {
  const int rr = lane & 15, sel = lane >> 4;
  const _Float16* ap0 = aBuf + rr * SA + sel * 8;
  const int NC = CT * 16;
  for (int c = 0; c < CT; ++c) {
    v8f acc = {};
    const _Float16* wp = wFrag + (c * KT) * 512 + lane * 16;
    for (int kt = 0; kt < KT; ++kt) {
      v16h a = ld_afrag(ap0 + kt * 32);
      v16h w = ld_bfrag(wp);
      wp += 512;
      acc = __builtin_amdgcn_wmma_f32_16x16x32_f16(false, a, false, w, (short)0, acc,
                                                   false, false);
    }
    const int col = c * 16 + rr;
#pragma unroll
    for (int r = 0; r < 8; ++r) hBuf[(r + sel * 8) * NC + col] = acc[r];
  }
}

// ---- relu -> LayerNorm over 128 features, emit f16 A for next layer -------
__device__ __forceinline__ void relu_ln_128(const float* __restrict__ hBuf,
                                            const float* __restrict__ bias,
                                            const float* __restrict__ gam,
                                            const float* __restrict__ bet,
                                            _Float16* __restrict__ aNext, int lane) {
  const int row = lane >> 1;
  const int base = (lane & 1) * 64;
  const float* hp = hBuf + row * HD + base;
  float s = 0.f, sq = 0.f;
#pragma unroll 4
  for (int k = 0; k < 64; k += 4) {
    float4 v  = *(const float4*)(hp + k);
    float4 bb = *(const float4*)(bias + base + k);
    float a0 = fmaxf(v.x + bb.x, 0.f), a1 = fmaxf(v.y + bb.y, 0.f);
    float a2 = fmaxf(v.z + bb.z, 0.f), a3 = fmaxf(v.w + bb.w, 0.f);
    s += a0 + a1 + a2 + a3;
    sq += a0 * a0 + a1 * a1 + a2 * a2 + a3 * a3;
  }
  s  += __shfl_xor(s, 1, 32);
  sq += __shfl_xor(sq, 1, 32);
  const float mu  = s * (1.f / 128.f);
  const float inv = rsqrtf(sq * (1.f / 128.f) - mu * mu + LN_EPS);
  _Float16* op = aNext + row * HD + base;
#pragma unroll 4
  for (int k = 0; k < 64; k += 4) {
    float4 v  = *(const float4*)(hp + k);
    float4 bb = *(const float4*)(bias + base + k);
    float4 gg = *(const float4*)(gam + base + k);
    float4 tt = *(const float4*)(bet + base + k);
    h4 o;
    o.x = (_Float16)((fmaxf(v.x + bb.x, 0.f) - mu) * inv * gg.x + tt.x);
    o.y = (_Float16)((fmaxf(v.y + bb.y, 0.f) - mu) * inv * gg.y + tt.y);
    o.z = (_Float16)((fmaxf(v.z + bb.z, 0.f) - mu) * inv * gg.z + tt.z);
    o.w = (_Float16)((fmaxf(v.w + bb.w, 0.f) - mu) * inv * gg.w + tt.w);
    *(h4*)(op + k) = o;
  }
}

// ===========================================================================
// Kernel 0: zero the workspace aggregates
// ===========================================================================
__global__ void gnb_zero_ws(float* p, int n) {
  int i = blockIdx.x * blockDim.x + threadIdx.x;
  if (i < n) p[i] = 0.f;
}

// ===========================================================================
// Kernel 1: edge MLP (dominant).  Tile = 16 consecutive edges (same b,i).
// ===========================================================================
__global__ void gnb_edge_kernel(const float* __restrict__ x,
                                const float* __restrict__ ea,
                                const float* __restrict__ u,
                                const float* __restrict__ W1, const float* __restrict__ b1,
                                const float* __restrict__ g1, const float* __restrict__ t1,
                                const float* __restrict__ W2, const float* __restrict__ b2,
                                const float* __restrict__ g2, const float* __restrict__ t2,
                                const float* __restrict__ W3, const float* __restrict__ b3,
                                float* __restrict__ outEdge,
                                float* __restrict__ wsIncoming,
                                float* __restrict__ wsEdgeAgg) {
  extern __shared__ __align__(16) char smem[];
  const int tid = threadIdx.x, lane = tid & 31, wave = tid >> 5;

  const int KIN = 2 * ND + ED + GD;              // 224
  _Float16* wF1 = (_Float16*)smem;               // (7*8)*512
  _Float16* wF2 = wF1 + 56 * 512;                // (4*8)*512
  _Float16* wF3 = wF2 + 32 * 512;                // (4*4)*512
  float*    prm = (float*)(wF3 + 16 * 512);      // 832 f32

  stage_weights(W1, KIN, HD, wF1, tid, blockDim.x);
  stage_weights(W2, HD, HD, wF2, tid, blockDim.x);
  stage_weights(W3, HD, ED, wF3, tid, blockDim.x);
  copy_f32(b1, prm, 128, tid, blockDim.x);
  copy_f32(g1, prm + 128, 128, tid, blockDim.x);
  copy_f32(t1, prm + 256, 128, tid, blockDim.x);
  copy_f32(b2, prm + 384, 128, tid, blockDim.x);
  copy_f32(g2, prm + 512, 128, tid, blockDim.x);
  copy_f32(t2, prm + 640, 128, tid, blockDim.x);
  copy_f32(b3, prm + 768, 64, tid, blockDim.x);
  __syncthreads();

  const float* PB1 = prm;       const float* PG1 = prm + 128; const float* PT1 = prm + 256;
  const float* PB2 = prm + 384; const float* PG2 = prm + 512; const float* PT2 = prm + 640;
  const float* PB3 = prm + 768;

  char* wb = (char*)(prm + 832) + wave * (7168 + 8192);
  _Float16* aBuf = (_Float16*)wb;                // 16 x 224 f16 (reused as 16x128)
  float*    hBuf = (float*)(wb + 7168);          // 16 x 128 f32

  const int nSlots = gridDim.x * (blockDim.x >> 5);
  const int slot = blockIdx.x * (blockDim.x >> 5) + wave;
  const int TILES = B_ * N_ * N_ / 16;
  const int rr = lane & 15, sel = lane >> 4;

  for (int tile = slot; tile < TILES; tile += nSlots) {
    const int b = tile / (N_ * N_ / 16);
    int rem = tile - b * (N_ * N_ / 16);
    const int i = rem / (N_ / 16);
    const int j0 = (rem - i * (N_ / 16)) * 16;

    // ---- prefetch the next tile this wave will process --------------------
    {
      const int nt = tile + nSlots;
      if (nt < TILES) {
        const int nb = nt / (N_ * N_ / 16);
        int nrem = nt - nb * (N_ * N_ / 16);
        const int ni = nrem / (N_ / 16);
        const int nj0 = (nrem - ni * (N_ / 16)) * 16;
        const char* pea = (const char*)&ea[(((size_t)(nb * N_ + ni)) * N_ + nj0) * ED];
        __builtin_prefetch(pea + lane * 128, 0, 3);          // 4KB edge_attr tile
        if (lane < 16) {
          const char* pxj = (const char*)&x[(nb * N_ + nj0) * ND];
          __builtin_prefetch(pxj + lane * 256, 0, 3);        // 4KB x_j rows
        }
      }
    }

    // ---- stage A: [x_i(64) | x_j(64) | edge_attr(64) | u(32)], f16 --------
    // vectorized: float4 global loads, 8-byte packed f16 LDS stores
    const size_t eabase = (((size_t)(b * N_ + i)) * N_ + j0) * ED;
    for (int t = lane; t < 16 * 16; t += 32) {
      const int r = t >> 4, f4 = (t & 15) << 2;
      const float4 xv = *(const float4*)&x[(b * N_ + j0 + r) * ND + f4];
      const float4 ev = *(const float4*)&ea[eabase + (size_t)r * ED + f4];
      *(h4*)&aBuf[r * 224 + 64 + f4]  = cvt4(xv);
      *(h4*)&aBuf[r * 224 + 128 + f4] = cvt4(ev);
    }
    if (lane < 16) {                       // x_i broadcast: 16 lanes x float4
      const float4 v = *(const float4*)&x[(b * N_ + i) * ND + lane * 4];
      const h4 hv = cvt4(v);
#pragma unroll
      for (int r = 0; r < 16; ++r) *(h4*)&aBuf[r * 224 + lane * 4] = hv;
    } else if (lane < 24) {                // u broadcast: 8 lanes x float4
      const int f4 = (lane - 16) * 4;
      const float4 v = *(const float4*)&u[b * GD + f4];
      const h4 hv = cvt4(v);
#pragma unroll
      for (int r = 0; r < 16; ++r) *(h4*)&aBuf[r * 224 + 192 + f4] = hv;
    }
    WAVE_FENCE();

    gemm_tile(aBuf, 224, 7, wF1, 8, hBuf, lane);           // layer 1 (56 WMMA)
    WAVE_FENCE();
    relu_ln_128(hBuf, PB1, PG1, PT1, aBuf, lane);          // A2 = LN(relu(h1))
    WAVE_FENCE();
    gemm_tile(aBuf, 128, 4, wF2, 8, hBuf, lane);           // layer 2 (32 WMMA)
    WAVE_FENCE();
    relu_ln_128(hBuf, PB2, PG2, PT2, aBuf, lane);          // A3
    WAVE_FENCE();

    // ---- layer 3 (16 WMMA) with register epilogue -------------------------
    const _Float16* ap0 = aBuf + rr * 128 + sel * 8;
    for (int c = 0; c < 4; ++c) {
      v8f acc = {};
      const _Float16* wp = wF3 + (c * 4) * 512 + lane * 16;
#pragma unroll
      for (int kt = 0; kt < 4; ++kt) {
        v16h a = ld_afrag(ap0 + kt * 32);
        v16h w = ld_bfrag(wp);
        wp += 512;
        acc = __builtin_amdgcn_wmma_f32_16x16x32_f16(false, a, false, w, (short)0, acc,
                                                     false, false);
      }
      const int col = c * 16 + rr;
      const float bv = PB3[col];
      float csum = 0.f;
#pragma unroll
      for (int r = 0; r < 8; ++r) {
        const float o = acc[r] + bv;
        outEdge[eabase + (size_t)(r + sel * 8) * ED + col] = o;
        csum += o;
      }
      csum += __shfl_xor(csum, 16, 32);                    // lanes l / l+16 share col
      if (sel == 0) {
        atomicAdd(&wsIncoming[(b * N_ + i) * ED + col], csum);
        atomicAdd(&wsEdgeAgg[b * ED + col], csum);
      }
    }
  }
}

// ===========================================================================
// Kernel 2: node MLP.  Tile = 16 consecutive nodes (same b).
// ===========================================================================
__global__ void gnb_node_kernel(const float* __restrict__ x,
                                const float* __restrict__ u,
                                const float* __restrict__ wsIncoming,
                                const float* __restrict__ W1, const float* __restrict__ b1,
                                const float* __restrict__ g1, const float* __restrict__ t1,
                                const float* __restrict__ W2, const float* __restrict__ b2,
                                const float* __restrict__ g2, const float* __restrict__ t2,
                                const float* __restrict__ W3, const float* __restrict__ b3,
                                float* __restrict__ outX,
                                float* __restrict__ wsNodeAgg) {
  extern __shared__ __align__(16) char smem[];
  const int tid = threadIdx.x, lane = tid & 31, wave = tid >> 5;

  const int KIN = ND + ED + GD;                  // 160
  _Float16* wF1 = (_Float16*)smem;               // (5*8)*512
  _Float16* wF2 = wF1 + 40 * 512;                // (4*8)*512
  _Float16* wF3 = wF2 + 32 * 512;                // (4*4)*512
  float*    prm = (float*)(wF3 + 16 * 512);      // 832 f32

  stage_weights(W1, KIN, HD, wF1, tid, blockDim.x);
  stage_weights(W2, HD, HD, wF2, tid, blockDim.x);
  stage_weights(W3, HD, ND, wF3, tid, blockDim.x);
  copy_f32(b1, prm, 128, tid, blockDim.x);
  copy_f32(g1, prm + 128, 128, tid, blockDim.x);
  copy_f32(t1, prm + 256, 128, tid, blockDim.x);
  copy_f32(b2, prm + 384, 128, tid, blockDim.x);
  copy_f32(g2, prm + 512, 128, tid, blockDim.x);
  copy_f32(t2, prm + 640, 128, tid, blockDim.x);
  copy_f32(b3, prm + 768, 64, tid, blockDim.x);
  __syncthreads();

  const float* PB1 = prm;       const float* PG1 = prm + 128; const float* PT1 = prm + 256;
  const float* PB2 = prm + 384; const float* PG2 = prm + 512; const float* PT2 = prm + 640;
  const float* PB3 = prm + 768;

  char* wb = (char*)(prm + 832) + wave * (5120 + 8192);
  _Float16* aBuf = (_Float16*)wb;                // 16 x 160 f16
  float*    hBuf = (float*)(wb + 5120);          // 16 x 128 f32

  const int nSlots = gridDim.x * (blockDim.x >> 5);
  const int slot = blockIdx.x * (blockDim.x >> 5) + wave;
  const int TILES = B_ * N_ / 16;
  const int rr = lane & 15, sel = lane >> 4;

  for (int tile = slot; tile < TILES; tile += nSlots) {
    const int g0 = tile * 16;
    const int b = g0 / N_;

    // ---- stage A: [x(64) | incoming(64) | u(32)] --------------------------
    for (int t = lane; t < 16 * 16; t += 32) {
      const int r = t >> 4, f4 = (t & 15) << 2;
      const float4 xv = *(const float4*)&x[(g0 + r) * ND + f4];
      const float4 iv = *(const float4*)&wsIncoming[(g0 + r) * ED + f4];
      *(h4*)&aBuf[r * 160 + f4]      = cvt4(xv);
      *(h4*)&aBuf[r * 160 + 64 + f4] = cvt4(iv);
    }
    if (lane < 8) {                        // u broadcast: 8 lanes x float4
      const float4 v = *(const float4*)&u[b * GD + lane * 4];
      const h4 hv = cvt4(v);
#pragma unroll
      for (int r = 0; r < 16; ++r) *(h4*)&aBuf[r * 160 + 128 + lane * 4] = hv;
    }
    WAVE_FENCE();

    gemm_tile(aBuf, 160, 5, wF1, 8, hBuf, lane);
    WAVE_FENCE();
    relu_ln_128(hBuf, PB1, PG1, PT1, aBuf, lane);
    WAVE_FENCE();
    gemm_tile(aBuf, 128, 4, wF2, 8, hBuf, lane);
    WAVE_FENCE();
    relu_ln_128(hBuf, PB2, PG2, PT2, aBuf, lane);
    WAVE_FENCE();

    const _Float16* ap0 = aBuf + rr * 128 + sel * 8;
    for (int c = 0; c < 4; ++c) {
      v8f acc = {};
      const _Float16* wp = wF3 + (c * 4) * 512 + lane * 16;
#pragma unroll
      for (int kt = 0; kt < 4; ++kt) {
        v16h a = ld_afrag(ap0 + kt * 32);
        v16h w = ld_bfrag(wp);
        wp += 512;
        acc = __builtin_amdgcn_wmma_f32_16x16x32_f16(false, a, false, w, (short)0, acc,
                                                     false, false);
      }
      const int col = c * 16 + rr;
      const float bv = PB3[col];
      float csum = 0.f;
#pragma unroll
      for (int r = 0; r < 8; ++r) {
        const float o = acc[r] + bv;
        outX[(size_t)(g0 + r + sel * 8) * ND + col] = o;
        csum += o;
      }
      csum += __shfl_xor(csum, 16, 32);
      if (sel == 0) atomicAdd(&wsNodeAgg[b * ND + col], csum);
    }
  }
}

// ===========================================================================
// Kernel 3: global MLP.  8 rows padded to one 16-row tile; wave 0 computes.
// ===========================================================================
__global__ void gnb_glob_kernel(const float* __restrict__ u,
                                const float* __restrict__ wsNodeAgg,
                                const float* __restrict__ wsEdgeAgg,
                                const float* __restrict__ W1, const float* __restrict__ b1,
                                const float* __restrict__ g1, const float* __restrict__ t1,
                                const float* __restrict__ W2, const float* __restrict__ b2,
                                const float* __restrict__ g2, const float* __restrict__ t2,
                                const float* __restrict__ W3, const float* __restrict__ b3,
                                float* __restrict__ outU) {
  extern __shared__ __align__(16) char smem[];
  const int tid = threadIdx.x, lane = tid & 31, wave = tid >> 5;

  const int KIN = ND + ED + GD;                  // 160
  _Float16* wF1 = (_Float16*)smem;               // 40*512
  _Float16* wF2 = wF1 + 40 * 512;                // 32*512
  _Float16* wF3 = wF2 + 32 * 512;                // (4*2)*512
  float*    prm = (float*)(wF3 + 8 * 512);       // 800 f32

  stage_weights(W1, KIN, HD, wF1, tid, blockDim.x);
  stage_weights(W2, HD, HD, wF2, tid, blockDim.x);
  stage_weights(W3, HD, GD, wF3, tid, blockDim.x);
  copy_f32(b1, prm, 128, tid, blockDim.x);
  copy_f32(g1, prm + 128, 128, tid, blockDim.x);
  copy_f32(t1, prm + 256, 128, tid, blockDim.x);
  copy_f32(b2, prm + 384, 128, tid, blockDim.x);
  copy_f32(g2, prm + 512, 128, tid, blockDim.x);
  copy_f32(t2, prm + 640, 128, tid, blockDim.x);
  copy_f32(b3, prm + 768, 32, tid, blockDim.x);

  _Float16* aBuf = (_Float16*)(prm + 800);       // 16 x 160 f16
  float*    hBuf = (float*)((char*)aBuf + 5120); // 16 x 128 f32

  // stage A with all threads: rows 0..7 = [node_agg | edge_agg | u], 8..15 = 0
  for (int t = tid; t < 16 * 160; t += blockDim.x) {
    const int r = t / 160, f = t - r * 160;
    float v = 0.f;
    if (r < B_) {
      if (f < 64) v = wsNodeAgg[r * 64 + f];
      else if (f < 128) v = wsEdgeAgg[r * 64 + (f - 64)];
      else v = u[r * GD + (f - 128)];
    }
    aBuf[t] = (_Float16)v;
  }
  __syncthreads();

  if (wave != 0) return;

  gemm_tile(aBuf, 160, 5, wF1, 8, hBuf, lane);
  WAVE_FENCE();
  relu_ln_128(hBuf, prm, prm + 128, prm + 256, aBuf, lane);
  WAVE_FENCE();
  gemm_tile(aBuf, 128, 4, wF2, 8, hBuf, lane);
  WAVE_FENCE();
  relu_ln_128(hBuf, prm + 384, prm + 512, prm + 640, aBuf, lane);
  WAVE_FENCE();
  gemm_tile(aBuf, 128, 4, wF3, 2, hBuf, lane);   // 16x32 output into hBuf
  WAVE_FENCE();

  const float* PB3 = prm + 768;
  for (int t = lane; t < B_ * GD; t += 32) {
    const int r = t >> 5, col = t & 31;
    outU[r * GD + col] = hBuf[r * GD + col] + PB3[col];
  }
}

// ===========================================================================
// Host launcher
// ===========================================================================
extern "C" void kernel_launch(void* const* d_in, const int* in_sizes, int n_in,
                              void* d_out, int out_size, void* d_ws, size_t ws_size,
                              hipStream_t stream) {
  (void)in_sizes; (void)n_in; (void)out_size; (void)ws_size;

  const float* x  = (const float*)d_in[0];   // [8,256,64]
  const float* ea = (const float*)d_in[1];   // [8,256,256,64]
  const float* u  = (const float*)d_in[2];   // [8,32]
  // d_in[3]=batch_size, d_in[4]=num_nodes (compile-time constants here)
  const float* eW1 = (const float*)d_in[5],  *eB1 = (const float*)d_in[6];
  const float* eG1 = (const float*)d_in[7],  *eT1 = (const float*)d_in[8];
  const float* eW2 = (const float*)d_in[9],  *eB2 = (const float*)d_in[10];
  const float* eG2 = (const float*)d_in[11], *eT2 = (const float*)d_in[12];
  const float* eW3 = (const float*)d_in[13], *eB3 = (const float*)d_in[14];
  const float* nW1 = (const float*)d_in[15], *nB1 = (const float*)d_in[16];
  const float* nG1 = (const float*)d_in[17], *nT1 = (const float*)d_in[18];
  const float* nW2 = (const float*)d_in[19], *nB2 = (const float*)d_in[20];
  const float* nG2 = (const float*)d_in[21], *nT2 = (const float*)d_in[22];
  const float* nW3 = (const float*)d_in[23], *nB3 = (const float*)d_in[24];
  const float* gW1 = (const float*)d_in[25], *gB1 = (const float*)d_in[26];
  const float* gG1 = (const float*)d_in[27], *gT1 = (const float*)d_in[28];
  const float* gW2 = (const float*)d_in[29], *gB2 = (const float*)d_in[30];
  const float* gG2 = (const float*)d_in[31], *gT2 = (const float*)d_in[32];
  const float* gW3 = (const float*)d_in[33], *gB3 = (const float*)d_in[34];

  float* out  = (float*)d_out;
  float* outX = out;                                     // 8*256*64      = 131072
  float* outE = out + 131072;                            // 8*256*256*64  = 33554432
  float* outU = out + 131072 + 33554432;                 // 8*32          = 256

  float* ws     = (float*)d_ws;
  float* wsInc  = ws;                 // [8*256, 64] incoming
  float* wsEA   = ws + 131072;        // [8, 64] edge_agg
  float* wsNA   = ws + 131584;        // [8, 64] node_agg
  const int WS_N = 132096;

  gnb_zero_ws<<<(WS_N + 255) / 256, 256, 0, stream>>>(ws, WS_N);

  // edge: weights 104 frags (106496B) + params (3328B) + 8 waves * 15360B
  const size_t lds_edge = 106496 + 3328 + 8 * (7168 + 8192);     // 232704
  gnb_edge_kernel<<<256, 256, lds_edge, stream>>>(
      x, ea, u, eW1, eB1, eG1, eT1, eW2, eB2, eG2, eT2, eW3, eB3,
      outE, wsInc, wsEA);

  // node: weights 88 frags (90112B) + params + 8 waves * 13312B
  const size_t lds_node = 90112 + 3328 + 8 * (5120 + 8192);      // 199936
  gnb_node_kernel<<<16, 256, lds_node, stream>>>(
      x, u, wsInc, nW1, nB1, nG1, nT1, nW2, nB2, nG2, nT2, nW3, nB3,
      outX, wsNA);

  // global: weights 80 frags (81920B) + params (3200B) + 1 wave buffers
  const size_t lds_glob = 81920 + 3200 + (5120 + 8192);          // 98432
  gnb_glob_kernel<<<1, 256, lds_glob, stream>>>(
      u, wsNA, wsEA, gW1, gB1, gG1, gT1, gW2, gB2, gG2, gT2, gW3, gB3,
      outU);
}
